// DocREModel_14920716387141
// MI455X (gfx1250) — compile-verified, matrix operand order checked by reading, not run on hardware
//
#include <hip/hip_runtime.h>
#include <hip/hip_bf16.h>
#include <math.h>

typedef __attribute__((ext_vector_type(16))) __bf16 v16bf;
typedef __attribute__((ext_vector_type(8)))  __bf16 v8bf;
typedef __attribute__((ext_vector_type(8)))  float  v8f;

constexpr int NB = 4, CC = 1024, DD = 768, HH = 12, EE = 24, MM = 4, PP = 552, RR = 97;
constexpr float INV_SQRT_D = 0.03608439182435161f; // 1/sqrt(768)

// =====================================================================
// bf16 WMMA GEMM:  C[m,n] = act( sum_k A(m,k) * Bt(n,k) + bias[n] )
// A  : bf16, row-major M x K (row stride sAm, contiguous k)
// Bt : bf16, row-major N x K (row stride sBn, contiguous k)  [pre-transposed B]
// One wave per block computes a 32x32 output tile (2x2 register blocking of
// 16x16 WMMA tiles): per K-step, 8 x global_load_b128 feed 4 x v_wmma
// (each A/B fragment reused twice). K must be a multiple of 32.
// Edge rows/cols are clamped (never stored), so the fragment path is
// branch-free and EXEC is all-ones around every v_wmma.
// =====================================================================
__global__ __launch_bounds__(32)
void wmma_gemm_bf(const __bf16* __restrict__ A, const __bf16* __restrict__ Bt,
                  float* __restrict__ Cf, __bf16* __restrict__ Cb,
                  int Mr, int Nc, int Kd,
                  int sAm, long sAb, int sBn, long sBb,
                  int sCm, int sCn, long sCb,
                  const float* __restrict__ bias, int actTanh)
{
    const int lane  = threadIdx.x & 31;
    const int half  = lane >> 4;          // 0: K 0..7/16..23, 1: K 8..15/24..31
    const int id16  = lane & 15;
    const int kbase = half * 8;
    const long b    = blockIdx.z;

    A  += b * sAb;
    Bt += b * sBb;

    int am0 = blockIdx.y * 32 + id16;       am0 = (am0 < Mr) ? am0 : (Mr - 1);
    int am1 = blockIdx.y * 32 + 16 + id16;  am1 = (am1 < Mr) ? am1 : (Mr - 1);
    int bn0 = blockIdx.x * 32 + id16;       bn0 = (bn0 < Nc) ? bn0 : (Nc - 1);
    int bn1 = blockIdx.x * 32 + 16 + id16;  bn1 = (bn1 < Nc) ? bn1 : (Nc - 1);

    const __bf16* Ap0 = A  + (long)am0 * sAm + kbase;
    const __bf16* Ap1 = A  + (long)am1 * sAm + kbase;
    const __bf16* Bp0 = Bt + (long)bn0 * sBn + kbase;
    const __bf16* Bp1 = Bt + (long)bn1 * sBn + kbase;

    v8f acc00 = {}, acc01 = {}, acc10 = {}, acc11 = {};
#pragma unroll 2
    for (int k0 = 0; k0 < Kd; k0 += 32) {
        union { v16bf v; v8bf h[2]; } a0, a1, b0, b1;
        a0.h[0] = *(const v8bf*)(Ap0);      a0.h[1] = *(const v8bf*)(Ap0 + 16);
        a1.h[0] = *(const v8bf*)(Ap1);      a1.h[1] = *(const v8bf*)(Ap1 + 16);
        b0.h[0] = *(const v8bf*)(Bp0);      b0.h[1] = *(const v8bf*)(Bp0 + 16);
        b1.h[0] = *(const v8bf*)(Bp1);      b1.h[1] = *(const v8bf*)(Bp1 + 16);
        acc00 = __builtin_amdgcn_wmma_f32_16x16x32_bf16(false, a0.v, false, b0.v,
                                                        (short)0, acc00, false, false);
        acc01 = __builtin_amdgcn_wmma_f32_16x16x32_bf16(false, a0.v, false, b1.v,
                                                        (short)0, acc01, false, false);
        acc10 = __builtin_amdgcn_wmma_f32_16x16x32_bf16(false, a1.v, false, b0.v,
                                                        (short)0, acc10, false, false);
        acc11 = __builtin_amdgcn_wmma_f32_16x16x32_bf16(false, a1.v, false, b1.v,
                                                        (short)0, acc11, false, false);
        Ap0 += 32; Ap1 += 32; Bp0 += 32; Bp1 += 32;
    }

    // Store 2x2 subtiles. D layout: lane -> N (id16), VGPR r -> M = r + 8*half.
    v8f accs[2][2] = { { acc00, acc01 }, { acc10, acc11 } };
#pragma unroll
    for (int ns = 0; ns < 2; ++ns) {
        const int n = blockIdx.x * 32 + ns * 16 + id16;
        if (n >= Nc) continue;
        const float bsv = bias ? bias[n] : 0.0f;
#pragma unroll
        for (int ms = 0; ms < 2; ++ms) {
            const int mbase = blockIdx.y * 32 + ms * 16 + half * 8;
#pragma unroll
            for (int r = 0; r < 8; ++r) {
                const int m = mbase + r;
                if (m < Mr) {
                    float v = accs[ms][ns][r] + bsv;
                    if (actTanh) v = tanhf(v);
                    const long off = b * sCb + (long)m * sCm + (long)n * sCn;
                    if (Cf) Cf[off] = v;
                    if (Cb) Cb[off] = (__bf16)v;
                }
            }
        }
    }
}

// ---------------- f32 -> bf16 straight copy ----------------
__global__ void cvt_bf16_k(const float* __restrict__ s, __bf16* __restrict__ d, int count)
{
    int i = blockIdx.x * blockDim.x + threadIdx.x;
    if (i < count) d[i] = (__bf16)s[i];
}

// ---------------- f32 (K x N) -> bf16 transposed (N x K), tiled ----------------
__global__ __launch_bounds__(256)
void transpose_cvt_k(const float* __restrict__ src, __bf16* __restrict__ dst,
                     int K, int N, long sbS, long sbD)
{
    __shared__ float tile[32][33];
    const long b = blockIdx.z;
    src += b * sbS; dst += b * sbD;
    const int k0 = blockIdx.x * 32, n0 = blockIdx.y * 32;
    const int tx = threadIdx.x, ty = threadIdx.y;
#pragma unroll
    for (int i = ty; i < 32; i += 8) {
        int k = k0 + i, n = n0 + tx;
        tile[i][tx] = (k < K && n < N) ? src[(long)k * N + n] : 0.0f;
    }
    __syncthreads();
#pragma unroll
    for (int i = ty; i < 32; i += 8) {
        int n = n0 + i, k = k0 + tx;
        if (n < N && k < K) dst[(long)n * K + k] = (__bf16)tile[tx][i];
    }
}

// ---------------- block reduction helpers ----------------
__device__ __forceinline__ float blk_sum(float v, float* red, int nt) {
    int t = threadIdx.x;
    red[t] = v; __syncthreads();
    for (int s = nt >> 1; s > 0; s >>= 1) { if (t < s) red[t] += red[t + s]; __syncthreads(); }
    float r = red[0]; __syncthreads(); return r;
}
__device__ __forceinline__ float blk_max(float v, float* red, int nt) {
    int t = threadIdx.x;
    red[t] = v; __syncthreads();
    for (int s = nt >> 1; s > 0; s >>= 1) { if (t < s) red[t] = fmaxf(red[t], red[t + s]); __syncthreads(); }
    float r = red[0]; __syncthreads(); return r;
}

// ---------------- K1: ent_emb = logsumexp over M mentions ----------------
__global__ void ent_emb_k(const float* __restrict__ seq, const int* __restrict__ epos,
                          float* __restrict__ ent_emb)
{
    int idx = blockIdx.x * blockDim.x + threadIdx.x;
    const int tot = NB * EE * DD;
    if (idx >= tot) return;
    int d = idx % DD, ne = idx / DD;
    int e = ne % EE,  n  = ne / EE;
    float vals[MM], mx = -3.0e38f;
#pragma unroll
    for (int m = 0; m < MM; ++m) {
        int pos = epos[(n * EE + e) * MM + m] + 1;
        float v = seq[((long)n * CC + pos) * DD + d];
        vals[m] = v; mx = fmaxf(mx, v);
    }
    float s = 0.0f;
#pragma unroll
    for (int m = 0; m < MM; ++m) s += expf(vals[m] - mx);
    ent_emb[idx] = mx + logf(s);
}

// ---------------- K2: ent_att = mean over M of gathered attention rows -----------
__global__ void ent_att_k(const float* __restrict__ att, const int* __restrict__ epos,
                          float* __restrict__ ent_att)
{
    int idx = blockIdx.x * blockDim.x + threadIdx.x;
    const int tot = NB * EE * HH * CC;
    if (idx >= tot) return;
    int c = idx % CC, r = idx / CC;
    int h = r % HH;   r /= HH;
    int e = r % EE;
    int n = r / EE;
    float s = 0.0f;
#pragma unroll
    for (int m = 0; m < MM; ++m) {
        int pos = epos[(n * EE + e) * MM + m] + 1;
        s += att[(((long)n * HH + h) * CC + pos) * CC + c];
    }
    ent_att[idx] = s * (1.0f / MM);
}

// ---------------- K3: ht_att (mean over heads, normalized over C) -> bf16 --------
__global__ __launch_bounds__(256)
void ht_att_k(const float* __restrict__ ent_att, const int* __restrict__ hts,
              __bf16* __restrict__ ht_att_bf)
{
    __shared__ float red[256];
    const int p = blockIdx.x, n = blockIdx.y;
    const int np = n * PP + p;
    const int hi = hts[np * 2], ti = hts[np * 2 + 1];
    const float* ha = ent_att + ((long)(n * EE + hi)) * HH * CC;
    const float* ta = ent_att + ((long)(n * EE + ti)) * HH * CC;
    float loc[CC / 256];
    float lsum = 0.0f;
#pragma unroll
    for (int i = 0; i < CC / 256; ++i) {
        int c = threadIdx.x + i * 256;
        float s = 0.0f;
        for (int h = 0; h < HH; ++h) s += ha[h * CC + c] * ta[h * CC + c];
        s *= (1.0f / HH);
        loc[i] = s; lsum += s;
    }
    float tot = blk_sum(lsum, red, 256);
    float inv = 1.0f / (tot + 1e-5f);
#pragma unroll
    for (int i = 0; i < CC / 256; ++i)
        ht_att_bf[(long)np * CC + threadIdx.x + i * 256] = (__bf16)(loc[i] * inv);
}

// ---------------- K5: gather hs/ts into htcat_bf columns [0,2D) ------------------
__global__ void gather_ht_k(const float* __restrict__ ent_emb, const int* __restrict__ hts,
                            __bf16* __restrict__ htcat_bf)
{
    int idx = blockIdx.x * blockDim.x + threadIdx.x;
    const int tot = NB * PP * DD;
    if (idx >= tot) return;
    int d = idx % DD, np = idx / DD;
    int n = np / PP;
    int hi = hts[np * 2], ti = hts[np * 2 + 1];
    htcat_bf[(long)np * 3 * DD + d]      = (__bf16)ent_emb[((long)n * EE + hi) * DD + d];
    htcat_bf[(long)np * 3 * DD + DD + d] = (__bf16)ent_emb[((long)n * EE + ti) * DD + d];
}

// ---------------- K10: rel-attention (scores->softmax->@v), per (h,n,p) ----------
__global__ __launch_bounds__(128)
void attn1_k(const float* __restrict__ q, const float* __restrict__ k,
             const float* __restrict__ v, float* __restrict__ out1)
{
    __shared__ float qv[64];
    __shared__ float sc[RR];
    __shared__ float red[128];
    const int p = blockIdx.x, n = blockIdx.y, h = blockIdx.z;
    const int np = n * PP + p;
    const int t = threadIdx.x;
    if (t < 64) qv[t] = q[(long)np * DD + h * 64 + t];
    __syncthreads();
    float s = -3.0e38f;
    if (t < RR) {
        float acc = 0.0f;
        const float* kr = k + (long)t * DD + h * 64;
        for (int e = 0; e < 64; ++e) acc += qv[e] * kr[e];
        s = acc * INV_SQRT_D;
        sc[t] = s;
    }
    float mx = blk_max(s, red, 128);
    float ev = 0.0f;
    if (t < RR) { ev = expf(sc[t] - mx); sc[t] = ev; }
    float tot = blk_sum(ev, red, 128);
    float inv = 1.0f / tot;
    __syncthreads();
    if (t < 64) {
        float o = 0.0f;
        const float* vc = v + h * 64 + t;
        for (int r = 0; r < RR; ++r) o += sc[r] * vc[(long)r * DD];
        out1[(long)np * DD + h * 64 + t] = o * inv;
    }
}

// ---------------- K11: pairwise self-attention (552x552 per head) ---------------
__global__ __launch_bounds__(256)
void attn2_k(const float* __restrict__ out1, float* __restrict__ out2)
{
    __shared__ float av[64];
    __shared__ float sc[PP];
    __shared__ float red[256];
    const int p = blockIdx.x, n = blockIdx.y, h = blockIdx.z;
    const int np = n * PP + p;
    const int t = threadIdx.x;
    if (t < 64) av[t] = out1[(long)np * DD + h * 64 + t];
    __syncthreads();
    for (int qq = t; qq < PP; qq += 256) {
        const float* o = out1 + ((long)(n * PP + qq)) * DD + h * 64;
        float s = 0.0f;
        for (int e = 0; e < 64; ++e) s += av[e] * o[e];
        sc[qq] = s * INV_SQRT_D;
    }
    __syncthreads();
    float lmax = -3.0e38f;
    for (int qq = t; qq < PP; qq += 256) lmax = fmaxf(lmax, sc[qq]);
    float mx = blk_max(lmax, red, 256);
    float lsum = 0.0f;
    for (int qq = t; qq < PP; qq += 256) { float e = expf(sc[qq] - mx); sc[qq] = e; lsum += e; }
    float tot = blk_sum(lsum, red, 256);
    float inv = 1.0f / tot;
    __syncthreads();
    if (t < 64) {
        float o = 0.0f;
        const float* vc = out1 + (long)n * PP * DD + h * 64 + t;
        for (int qq = 0; qq < PP; ++qq) o += sc[qq] * vc[(long)qq * DD];
        out2[(long)np * DD + h * 64 + t] = o * inv;
    }
}

// ---------------- K12: x_bf = LN(htr + tanh(out2)) ----------------
__global__ __launch_bounds__(256)
void ln_k(const float* __restrict__ htr, const float* __restrict__ out2,
          const float* __restrict__ g, const float* __restrict__ bta,
          __bf16* __restrict__ x_bf)
{
    __shared__ float red[256];
    const long np = blockIdx.x;
    const int t = threadIdx.x;
    float xv[DD / 256];
    float lsum = 0.0f;
#pragma unroll
    for (int i = 0; i < DD / 256; ++i) {
        int d = t + i * 256;
        float vv = htr[np * DD + d] + tanhf(out2[np * DD + d]);
        xv[i] = vv; lsum += vv;
    }
    float mu = blk_sum(lsum, red, 256) * (1.0f / DD);
    float lv = 0.0f;
#pragma unroll
    for (int i = 0; i < DD / 256; ++i) { float dfl = xv[i] - mu; lv += dfl * dfl; }
    float var = blk_sum(lv, red, 256) * (1.0f / DD);
    float rstd = rsqrtf(var + 1e-5f);
#pragma unroll
    for (int i = 0; i < DD / 256; ++i) {
        int d = t + i * 256;
        x_bf[np * DD + d] = (__bf16)((xv[i] - mu) * rstd * g[d] + bta[d]);
    }
}

// ---------------- host side ----------------
extern "C" void kernel_launch(void* const* d_in, const int* in_sizes, int n_in,
                              void* d_out, int out_size, void* d_ws, size_t ws_size,
                              hipStream_t stream)
{
    const float* seq  = (const float*)d_in[0];
    const float* att  = (const float*)d_in[1];
    const int*   epos = (const int*)  d_in[2];
    const int*   hts  = (const int*)  d_in[3];
    const float* rels = (const float*)d_in[4];
    const float* We   = (const float*)d_in[5];
    const float* be   = (const float*)d_in[6];
    const float* Wq   = (const float*)d_in[7];
    const float* bq   = (const float*)d_in[8];
    const float* Wk   = (const float*)d_in[9];
    const float* bk   = (const float*)d_in[10];
    const float* Wv   = (const float*)d_in[11];
    const float* bv   = (const float*)d_in[12];
    const float* ln_g = (const float*)d_in[13];
    const float* ln_b = (const float*)d_in[14];
    const float* Wc   = (const float*)d_in[15];
    const float* bc   = (const float*)d_in[16];
    float* out = (float*)d_out;
    float* ws  = (float*)d_ws;

    const int NP = NB * PP;              // 2208

    // ---- f32 workspace (floats) ----
    float* ent_emb = ws;                 //    73,728  (N,E,D)
    float* ent_att = ws + 73728;         // 1,179,648  (N,E,H,C)
    float* htr     = ws + 1253376;       // 1,695,744  (N*P, D)
    float* qbuf    = ws + 2949120;       // 1,695,744  (N*P, D)
    float* kbuf    = ws + 4644864;       //    74,496  (R, D)
    float* vbuf    = ws + 4719360;       //    74,496  (R, D)
    float* out1    = ws + 4793856;       // 1,695,744  (N*P, D)
    float* out2    = qbuf;               // reuse: qbuf dead after attn1
    // ---- bf16 workspace (16B aligned: starts at float offset 6,489,600) ----
    __bf16* bfbase   = (__bf16*)(ws + 6489600);
    __bf16* ht_attB  = bfbase;            // 2,260,992  (N,P,C)
    __bf16* seqT     = bfbase + 2260992;  // 3,145,728  (N, D, C)  seq transposed
    __bf16* htcatB   = bfbase + 5406720;  // 5,087,232  (N*P, 3D)
    __bf16* WeT      = bfbase + 10493952; // 1,769,472  (D, 3D)
    __bf16* htrB     = bfbase + 12263424; // 1,695,744  (N*P, D)
    __bf16* WqT      = bfbase + 13959168; //   589,824  (D, D)
    __bf16* WkT      = bfbase + 14548992; //   589,824
    __bf16* WvT      = bfbase + 15138816; //   589,824
    __bf16* relsB    = bfbase + 15728640; //    74,496  (R, D)
    __bf16* xB       = bfbase + 15803136; // 1,695,744  (N*P, D)
    __bf16* WcT      = bfbase + 17498880; //    74,496  (R, D)   Wc transposed
    // total ws ~ 58.3 MB

    dim3 tb(32, 8);
    // ---- one-time operand conversions (independent; launched first) ----
    transpose_cvt_k<<<dim3(CC/32, DD/32, NB), tb, 0, stream>>>(seq, seqT, CC, DD,
                                                              (long)CC*DD, (long)DD*CC);
    transpose_cvt_k<<<dim3(3*DD/32, DD/32, 1), tb, 0, stream>>>(We, WeT, 3*DD, DD, 0, 0);
    transpose_cvt_k<<<dim3(DD/32, DD/32, 1), tb, 0, stream>>>(Wq, WqT, DD, DD, 0, 0);
    transpose_cvt_k<<<dim3(DD/32, DD/32, 1), tb, 0, stream>>>(Wk, WkT, DD, DD, 0, 0);
    transpose_cvt_k<<<dim3(DD/32, DD/32, 1), tb, 0, stream>>>(Wv, WvT, DD, DD, 0, 0);
    transpose_cvt_k<<<dim3(DD/32, (RR+31)/32, 1), tb, 0, stream>>>(Wc, WcT, DD, RR, 0, 0);
    cvt_bf16_k<<<(RR*DD + 255)/256, 256, 0, stream>>>(rels, relsB, RR*DD);

    // K1: ent_emb = logsumexp over mentions
    ent_emb_k<<<(NB*EE*DD + 255)/256, 256, 0, stream>>>(seq, epos, ent_emb);
    // K2: ent_att = mean over mentions of gathered attention rows
    ent_att_k<<<(NB*EE*HH*CC + 255)/256, 256, 0, stream>>>(att, epos, ent_att);
    // K3: ht_att (bf16)
    ht_att_k<<<dim3(PP, NB), 256, 0, stream>>>(ent_att, hts, ht_attB);
    // K5: gather hs/ts into htcatB[:, 0:2D)
    gather_ht_k<<<(NB*PP*DD + 255)/256, 256, 0, stream>>>(ent_emb, hts, htcatB);

    // K4: rs = ht_att @ seq  -> htcatB[:, 2D:3D)   (batched over n; bf16-only out)
    wmma_gemm_bf<<<dim3(DD/32, (PP+31)/32, NB), 32, 0, stream>>>(
        ht_attB, seqT, nullptr, htcatB + 2*DD,
        PP, DD, CC,
        CC, (long)PP*CC,  CC, (long)DD*CC,
        3*DD, 1, (long)PP*3*DD,
        nullptr, 0);
    // K6: htr = tanh(htcat @ We + be)   (f32 + bf16 outs)
    wmma_gemm_bf<<<dim3(DD/32, NP/32, 1), 32, 0, stream>>>(
        htcatB, WeT, htr, htrB,
        NP, DD, 3*DD,
        3*DD, 0,  3*DD, 0,
        DD, 1, 0,
        be, 1);
    // K7: q = htr @ Wq + bq
    wmma_gemm_bf<<<dim3(DD/32, NP/32, 1), 32, 0, stream>>>(
        htrB, WqT, qbuf, nullptr, NP, DD, DD,
        DD, 0,  DD, 0,  DD, 1, 0,  bq, 0);
    // K8/K9: k = rels @ Wk + bk ; v = rels @ Wv + bv
    wmma_gemm_bf<<<dim3(DD/32, (RR+31)/32, 1), 32, 0, stream>>>(
        relsB, WkT, kbuf, nullptr, RR, DD, DD,
        DD, 0,  DD, 0,  DD, 1, 0,  bk, 0);
    wmma_gemm_bf<<<dim3(DD/32, (RR+31)/32, 1), 32, 0, stream>>>(
        relsB, WvT, vbuf, nullptr, RR, DD, DD,
        DD, 0,  DD, 0,  DD, 1, 0,  bv, 0);
    // K10: rel-attention (fused softmax)
    attn1_k<<<dim3(PP, NB, HH), 128, 0, stream>>>(qbuf, kbuf, vbuf, out1);
    // K11: pairwise self-attention (fused softmax)
    attn2_k<<<dim3(PP, NB, HH), 256, 0, stream>>>(out1, out2);
    // K12: x = LayerNorm(htr + tanh(out2))  -> bf16
    ln_k<<<NP, 256, 0, stream>>>(htr, out2, ln_g, ln_b, xB);
    // K13: logits = x @ Wc + bc  -> d_out (f32)
    wmma_gemm_bf<<<dim3((RR+31)/32, NP/32, 1), 32, 0, stream>>>(
        xB, WcT, out, nullptr, NP, RR, DD,
        DD, 0,  DD, 0,  RR, 1, 0,  bc, 0);
}
